// Net_2405181686363
// MI455X (gfx1250) — matrix operand order repaired and checked by reading, not compiled, not run on hardware
//
#include <hip/hip_runtime.h>
#include <hip/hip_bf16.h>

// ---------------------------------------------------------------------------
// ECC GNN (2x edge-conditioned conv, CH=32) on gfx1250 using V_WMMA_F32_16X16X32_F16.
// Fusion: msg[e,:] = sum_s e[e,s]*(x_src @ W_s) + x_src @ B  (no [E,32,32] kernel tensor).
// ---------------------------------------------------------------------------

#define NN 100000   // nodes
#define NE 200000   // edges
#define NG 4096     // graphs
#define FIN 11
#define CH 32
#define NOUT 19

typedef __attribute__((ext_vector_type(16))) _Float16 v16h;
typedef __attribute__((ext_vector_type(8)))  _Float16 v8h;
typedef __attribute__((ext_vector_type(8)))  float    v8f;

__device__ __forceinline__ v8f wmma16(v16h a, v16h b, v8f c) {
    // emits v_wmma_f32_16x16x32_f16
    return __builtin_amdgcn_wmma_f32_16x16x32_f16(
        false, a, false, b, (short)0, c, false, false);
}

// A-operand (16x32 f16, M x K): lane L holds row M = L&15.
// lanes 0-15:  halfs 0..7 = K 0..7,  halfs 8..15 = K 16..23
// lanes 16-31: halfs 0..7 = K 8..15, halfs 8..15 = K 24..31
__device__ __forceinline__ v16h load_a_tile(const _Float16* __restrict__ rowp, int lane) {
    const int koff = (lane & 16) ? 8 : 0;
    v8h lo = *(const v8h*)(rowp + koff);
    v8h hi = *(const v8h*)(rowp + koff + 16);
    v16h a;
#pragma unroll
    for (int i = 0; i < 8; ++i) { a[i] = lo[i]; a[i + 8] = hi[i]; }
    return a;
}

// ---------------------------------------------------------------------------
// root transform: out[r,:] = in[r,:] @ rW + bias   (in f16 padded, out f32)
// B-operand (32x16 f16, K x N): lane L holds column N=L&15, K in the same order as load_a_tile.
// ---------------------------------------------------------------------------
__global__ __launch_bounds__(32)
void root_wmma_kernel(const _Float16* __restrict__ inh,   // [N,32] f16
                      const _Float16* __restrict__ rW,    // [32][32] f16
                      const float*    __restrict__ bias,  // [32]
                      float*          __restrict__ outp,  // [N,32] f32
                      int n_rows)
{
    const int lane = threadIdx.x;
    const int ntiles = (n_rows + 15) >> 4;
    v16h b0, b1;
#pragma unroll
    for (int i = 0; i < 8; ++i) {
        const int kb = (lane & 16) ? 8 : 0;
        b0[i]     = rW[(kb + i) * 32 + (lane & 15)];
        b0[i + 8] = rW[(kb + 16 + i) * 32 + (lane & 15)];
        b1[i]     = rW[(kb + i) * 32 + 16 + (lane & 15)];
        b1[i + 8] = rW[(kb + 16 + i) * 32 + 16 + (lane & 15)];
    }
    const int n  = lane & 15;
    const int mb = (lane >> 4) << 3;
    const float bn0 = bias[n], bn1 = bias[n + 16];

    for (int tile = blockIdx.x; tile < ntiles; tile += gridDim.x) {
        const int r0 = tile << 4;
        int row = r0 + (lane & 15);
        if (row >= n_rows) row = n_rows - 1;           // clamp; stores guarded below
        v16h a = load_a_tile(inh + (size_t)row * 32, lane);
        v8f c0 = {}, c1 = {};
        c0 = wmma16(a, b0, c0);
        c1 = wmma16(a, b1, c1);
        float* op = outp + (size_t)(r0 + mb) * 32 + n;
        if (r0 + 16 <= n_rows) {                       // full tile: branch-free stores
#pragma unroll
            for (int i = 0; i < 8; ++i) {
                op[i * 32]      = c0[i] + bn0;
                op[i * 32 + 16] = c1[i] + bn1;
            }
        } else {
#pragma unroll
            for (int i = 0; i < 8; ++i) {
                if (r0 + mb + i < n_rows) {
                    op[i * 32]      = c0[i] + bn0;
                    op[i * 32 + 16] = c1[i] + bn1;
                }
            }
        }
    }
}

// ---------------------------------------------------------------------------
// edge messages: per 16-edge tile, gather A = inh[src], z_s = A @ W_s via WMMA,
// msg = sum_s e_s*z_s + z_b, scatter-add into accum[edge_dst].
// B operands (5 weight sets x 2 N-halves) held in registers across tiles.
// ---------------------------------------------------------------------------
__global__ __launch_bounds__(32)
void edge_wmma_kernel(const _Float16* __restrict__ inh,   // [N,32] f16
                      const _Float16* __restrict__ Wc,    // [5][32][32] f16
                      const float*    __restrict__ efeat, // [E,4]
                      const int*      __restrict__ esrc,
                      const int*      __restrict__ edst,
                      float*          __restrict__ accum, // [N,32] f32
                      int n_edges)
{
    __shared__ float s_e[16][4];
    __shared__ int   s_dst[16];
    const int lane = threadIdx.x;
    const int ntiles = (n_edges + 15) >> 4;

    // hoist all 10 B operands into registers (reused across the tile loop)
    v16h B[5][2];
#pragma unroll
    for (int s = 0; s < 5; ++s)
#pragma unroll
        for (int h = 0; h < 2; ++h)
#pragma unroll
            for (int i = 0; i < 8; ++i) {
                B[s][h][i]     = Wc[(size_t)s * 1024 + (((lane & 16) ? 8 : 0) + i) * 32 + h * 16 + (lane & 15)];
                B[s][h][i + 8] = Wc[(size_t)s * 1024 + (((lane & 16) ? 8 : 0) + 16 + i) * 32 + h * 16 + (lane & 15)];
            }

    const int n  = lane & 15;
    const int mb = (lane >> 4) << 3;

    for (int tile = blockIdx.x; tile < ntiles; tile += gridDim.x) {
        const int e0 = tile << 4;
        const bool full = (e0 + 16 <= n_edges);

        // prefetch next tile's edge metadata (global_prefetch_b8)
        const int ptile = tile + gridDim.x;
        if (ptile < ntiles) {
            const int pe = ptile << 4;
            __builtin_prefetch(esrc + pe, 0, 0);
            __builtin_prefetch(edst + pe, 0, 0);
            __builtin_prefetch(efeat + (size_t)pe * 4, 0, 0);
        }

        if (lane < 16) {
            const int ei = e0 + lane;
            const bool ok = ei < n_edges;
            const int ec = ok ? ei : 0;
            s_dst[lane] = ok ? edst[ec] : -1;
            float4 ev = ((const float4*)efeat)[ec];
            s_e[lane][0] = ev.x; s_e[lane][1] = ev.y;
            s_e[lane][2] = ev.z; s_e[lane][3] = ev.w;
        }
        __syncthreads();

        const int mi  = e0 + (lane & 15);
        const int src = (mi < n_edges) ? esrc[mi] : 0;
        v16h a = load_a_tile(inh + (size_t)src * 32, lane);

#pragma unroll
        for (int half = 0; half < 2; ++half) {
            v8f z0 = {}, z1 = {}, z2 = {}, z3 = {}, zb = {};
            z0 = wmma16(a, B[0][half], z0);
            z1 = wmma16(a, B[1][half], z1);
            z2 = wmma16(a, B[2][half], z2);
            z3 = wmma16(a, B[3][half], z3);
            zb = wmma16(a, B[4][half], zb);
            if (full) {
#pragma unroll
                for (int v = 0; v < 8; ++v) {
                    const int m = mb + v;
                    const int dst = s_dst[m];
                    float msg = s_e[m][0] * z0[v] + s_e[m][1] * z1[v]
                              + s_e[m][2] * z2[v] + s_e[m][3] * z3[v] + zb[v];
                    atomicAdd(accum + (size_t)dst * 32 + half * 16 + n, msg);
                }
            } else {
#pragma unroll
                for (int v = 0; v < 8; ++v) {
                    const int m = mb + v;
                    const int dst = s_dst[m];
                    if (dst >= 0) {
                        float msg = s_e[m][0] * z0[v] + s_e[m][1] * z1[v]
                                  + s_e[m][2] * z2[v] + s_e[m][3] * z3[v] + zb[v];
                        atomicAdd(accum + (size_t)dst * 32 + half * 16 + n, msg);
                    }
                }
            }
        }
        __syncthreads();   // WAR on s_e/s_dst before next tile's staging
    }
}

// ---------------------------------------------------------------------------
// helper kernels
// ---------------------------------------------------------------------------
__global__ void build_weights_kernel(const float* __restrict__ net1_W, const float* __restrict__ net1_b,
                                     const float* __restrict__ root1,
                                     const float* __restrict__ net2_W, const float* __restrict__ net2_b,
                                     const float* __restrict__ root2,
                                     _Float16* __restrict__ W1c, _Float16* __restrict__ root1h,
                                     _Float16* __restrict__ W2c, _Float16* __restrict__ root2h)
{
    int i = blockIdx.x * blockDim.x + threadIdx.x;
    if (i < 5120) {                       // W1c[5][32][32]: K-padded 11->32
        int s = i >> 10, k = (i >> 5) & 31, n = i & 31;
        float v = 0.f;
        if (k < FIN) v = (s < 4) ? net1_W[s * (FIN * CH) + k * CH + n]
                                 : net1_b[k * CH + n];
        W1c[i] = (_Float16)v;
    } else if (i < 6144) {                // root1h[32][32]: K-padded
        int j = i - 5120; int k = j >> 5, n = j & 31;
        root1h[j] = (_Float16)((k < FIN) ? root1[k * CH + n] : 0.f);
    } else if (i < 11264) {               // W2c[5][32][32]
        int j = i - 6144; int s = j >> 10, k = (j >> 5) & 31, n = j & 31;
        W2c[j] = (_Float16)((s < 4) ? net2_W[s * (CH * CH) + k * CH + n]
                                    : net2_b[k * CH + n]);
    } else if (i < 12288) {               // root2h[32][32]
        int j = i - 11264;
        root2h[j] = (_Float16)root2[j];
    }
}

__global__ void pad_x_kernel(const float* __restrict__ x, _Float16* __restrict__ xh, int n_rows) {
    int i = blockIdx.x * blockDim.x + threadIdx.x;
    if (i >= n_rows * 32) return;
    int n = i >> 5, k = i & 31;
    xh[i] = (_Float16)((k < FIN) ? x[n * FIN + k] : 0.f);
}

__global__ void relu_to_h16_kernel(const float* __restrict__ in, _Float16* __restrict__ out, int count) {
    int i = blockIdx.x * blockDim.x + threadIdx.x;
    if (i < count) { float v = in[i]; out[i] = (_Float16)(v > 0.f ? v : 0.f); }
}

__global__ void relu_f32_kernel(const float* __restrict__ in, float* __restrict__ out, int count) {
    int i = blockIdx.x * blockDim.x + threadIdx.x;
    if (i < count) { float v = in[i]; out[i] = v > 0.f ? v : 0.f; }
}

__global__ void zero_kernel(float* __restrict__ p, int count) {
    int i = blockIdx.x * blockDim.x + threadIdx.x;
    if (i < count) p[i] = 0.f;
}

// graph pooling: graph_idx is sorted, so run-length accumulate 8 consecutive
// nodes per lane (lane = channel) and emit one atomic per graph-run.
__global__ void pool_kernel(const float* __restrict__ h, const int* __restrict__ gidx,
                            float* __restrict__ pooled, int n_rows) {
    const int lane = threadIdx.x & 31;
    const int wave = (blockIdx.x * blockDim.x + threadIdx.x) >> 5;
    const int n0 = wave * 8;
    if (n0 >= n_rows) return;
    int cur_g = gidx[n0];
    float acc = 0.f;
#pragma unroll
    for (int i = 0; i < 8; ++i) {
        const int nrow = n0 + i;
        if (nrow >= n_rows) break;
        const int g = gidx[nrow];
        if (g != cur_g) {
            atomicAdd(pooled + (size_t)cur_g * 32 + lane, acc);
            cur_g = g; acc = 0.f;
        }
        acc += h[(size_t)nrow * 32 + lane];
    }
    atomicAdd(pooled + (size_t)cur_g * 32 + lane, acc);
}

__global__ void dense_kernel(const float* __restrict__ pooled, const float* __restrict__ dW,
                             const float* __restrict__ db, float* __restrict__ out) {
    int i = blockIdx.x * blockDim.x + threadIdx.x;
    if (i >= NG * NOUT) return;
    int g = i / NOUT, o = i % NOUT;
    float acc = db[o];
#pragma unroll
    for (int c = 0; c < CH; ++c) acc += pooled[g * CH + c] * dW[c * NOUT + o];
    out[i] = acc;
}

// ---------------------------------------------------------------------------
extern "C" void kernel_launch(void* const* d_in, const int* in_sizes, int n_in,
                              void* d_out, int out_size, void* d_ws, size_t ws_size,
                              hipStream_t stream) {
    const float* x       = (const float*)d_in[0];
    const float* e       = (const float*)d_in[1];
    const float* net1_W  = (const float*)d_in[2];
    const float* net1_b  = (const float*)d_in[3];
    const float* root1   = (const float*)d_in[4];
    const float* b1      = (const float*)d_in[5];
    const float* net2_W  = (const float*)d_in[6];
    const float* net2_b  = (const float*)d_in[7];
    const float* root2   = (const float*)d_in[8];
    const float* b2      = (const float*)d_in[9];
    const float* dense_W = (const float*)d_in[10];
    const float* dense_b = (const float*)d_in[11];
    const int* edge_src  = (const int*)d_in[12];
    const int* edge_dst  = (const int*)d_in[13];
    const int* graph_idx = (const int*)d_in[14];
    float* out = (float*)d_out;

    char* ws = (char*)d_ws;
    size_t off = 0;
    auto carve = [&](size_t bytes) -> void* {
        void* p = ws + off;
        off = (off + bytes + 255) & ~(size_t)255;
        return p;
    };
    _Float16* xh     = (_Float16*)carve((size_t)NN * 32 * 2);   // padded x, f16
    float*    h_pre  = (float*)   carve((size_t)NN * 32 * 4);   // pre-activation accumulator
    _Float16* h1h    = (_Float16*)carve((size_t)NN * 32 * 2);   // relu(h1) f16
    float*    h2     = (float*)   carve((size_t)NN * 32 * 4);   // relu(h2) f32
    float*    pooled = (float*)   carve((size_t)NG * 32 * 4);
    _Float16* W1c    = (_Float16*)carve(5 * 32 * 32 * 2);
    _Float16* root1h = (_Float16*)carve(32 * 32 * 2);
    _Float16* W2c    = (_Float16*)carve(5 * 32 * 32 * 2);
    _Float16* root2h = (_Float16*)carve(32 * 32 * 2);

    const int nel     = NN * 32;
    const int nt256   = (nel + 255) / 256;
    const int ntilesN = (NN + 15) / 16;     // 6250
    const int ntilesE = (NE + 15) / 16;     // 12500
    const int gRoot   = ntilesN < 2048 ? ntilesN : 2048;   // grid-stride: reuse B regs
    const int gEdge   = ntilesE < 4096 ? ntilesE : 4096;

    // precompute f16 weights + padded input
    build_weights_kernel<<<(12288 + 255) / 256, 256, 0, stream>>>(
        net1_W, net1_b, root1, net2_W, net2_b, root2, W1c, root1h, W2c, root2h);
    pad_x_kernel<<<nt256, 256, 0, stream>>>(x, xh, NN);

    // ---- layer 1 ----
    root_wmma_kernel<<<gRoot, 32, 0, stream>>>(xh, root1h, b1, h_pre, NN);
    edge_wmma_kernel<<<gEdge, 32, 0, stream>>>(xh, W1c, e, edge_src, edge_dst, h_pre, NE);
    relu_to_h16_kernel<<<nt256, 256, 0, stream>>>(h_pre, h1h, nel);

    // ---- layer 2 ----
    root_wmma_kernel<<<gRoot, 32, 0, stream>>>(h1h, root2h, b2, h_pre, NN);
    edge_wmma_kernel<<<gEdge, 32, 0, stream>>>(h1h, W2c, e, edge_src, edge_dst, h_pre, NE);
    relu_f32_kernel<<<nt256, 256, 0, stream>>>(h_pre, h2, nel);

    // ---- pooling + dense ----
    zero_kernel<<<(NG * 32 + 255) / 256, 256, 0, stream>>>(pooled, NG * 32);
    pool_kernel<<<(((NN + 7) / 8) * 32 + 255) / 256, 256, 0, stream>>>(h2, graph_idx, pooled, NN);
    dense_kernel<<<(NG * NOUT + 255) / 256, 256, 0, stream>>>(pooled, dense_W, dense_b, out);
}
